// CONCH_Finetuning_Model_TopjPooling_MoreText_10771777979154
// MI455X (gfx1250) — matrix operand order repaired and checked by reading, not compiled, not run on hardware
//
#include <hip/hip_runtime.h>
#include <hip/hip_bf16.h>
#include <math.h>

typedef __attribute__((ext_vector_type(16))) _Float16 v16h;
typedef __attribute__((ext_vector_type(8)))  float    v8f;

#define DDIM   512
#define KT_N   16         // K tiles of 32 -> 512
#define ROWS   16
#define NPB    50000
#define TILES  6250       // 100000 / 16
#define TOPJ   100
#define NCLS   5
#define NDESC  40
#define AP_TILE (KT_N*32*16)   // halves per packed 16-row tile (8192)

// workspace byte offsets
#define WS_WT    0                          // 512*512*2          = 524288
#define WS_DESC  524288                     // 16*3*32*16*2       = 49152
#define WS_PROJ  573440                     // 2*6250*8192*2      = 204800000
#define WS_PMAX  (573440 + 204800000)       // 100000*4           = 400000
#define WS_TOPI  (WS_PMAX + 400000)         // 2*128*4

// ---- WMMA f16 layout helpers (ISA 7.12.2) ----
// A 16x32: lane = M + 16*hi ; element e -> K = e + (e>=8 ? 8:0) + 8*hi
// B 32x16: lane = N%16 + 16*hi ; element e -> K = e + (e>=8 ? 8:0) + 8*hi
__device__ __forceinline__ int a_kk(int e, int lane) {
    return e + ((e >= 8) ? 8 : 0) + ((lane >= 16) ? 8 : 0);
}

// ---------------- pack W^T into WMMA-B fragments ----------------
__global__ void pack_w_kernel(const float* __restrict__ W, _Float16* __restrict__ wt) {
    int i = blockIdx.x * blockDim.x + threadIdx.x;          // ((kt*32+ct)*32+lane)*16+e
    if (i >= 512 * 512) return;
    int e = i & 15, lane = (i >> 4) & 31, ct = (i >> 9) & 31, kt = i >> 14;
    int k = kt * 32 + a_kk(e, lane);
    int j = ct * 16 + (lane & 15);
    wt[i] = (_Float16)W[j * DDIM + k];                      // B[k][j] = W[j][k]
}

__global__ void pack_desc_kernel(const float* __restrict__ Dt, _Float16* __restrict__ dp) {
    int i = blockIdx.x * blockDim.x + threadIdx.x;          // ((kt*3+ct)*32+lane)*16+e
    if (i >= 16 * 3 * 32 * 16) return;
    int e = i & 15, lane = (i >> 4) & 31;
    int rem = i >> 9, ct = rem % 3, kt = rem / 3;
    int k = kt * 32 + a_kk(e, lane);
    int t = ct * 16 + (lane & 15);
    dp[i] = (t < NDESC) ? (_Float16)Dt[t * DDIM + k] : (_Float16)0.0f;
}

// ---------------- projection GEMM + double l2norm ----------------
__global__ __launch_bounds__(256) void proj_gemm_kernel(
    const float* __restrict__ xs, const float* __restrict__ xl,
    const float* __restrict__ bias, const _Float16* __restrict__ wt,
    _Float16* __restrict__ proj)
{
    __shared__ float  xLds[ROWS * DDIM];   // 32KB (reused as epilogue buffer)
    __shared__ v16h   aLds[KT_N * 32];     // 16KB packed A fragments
    __shared__ float  red[256];
    __shared__ float  invn[ROWS];
    __shared__ float  oinv[ROWS];
    __shared__ float  bLds[DDIM];

    const int tile = blockIdx.x;
    const int mag  = blockIdx.y;
    const int tid  = threadIdx.x;
    const float* xsrc = (mag ? xl : xs) + (size_t)tile * ROWS * DDIM;

    // stage 16x512 fp32 rows + bias
    {
        const float4* s4 = (const float4*)xsrc;
        float4* d4 = (float4*)xLds;
        for (int v = tid; v < ROWS * DDIM / 4; v += 256) d4[v] = s4[v];
        for (int c = tid; c < DDIM; c += 256) bLds[c] = bias[c];
    }
    __syncthreads();

    // inner row norms (deferred normalization: l2norm(x)@Wt = (x@Wt)/||x||)
    {
        int row = tid >> 4, lp = tid & 15;
        float s = 0.f;
        for (int c = lp; c < DDIM; c += 16) { float f = xLds[row * DDIM + c]; s += f * f; }
        red[tid] = s;
        __syncthreads();
        if (lp == 0) {
            float t = 0.f;
            for (int q = 0; q < 16; ++q) t += red[(row << 4) + q];
            invn[row] = 1.0f / fmaxf(sqrtf(t), 1e-12f);
        }
        __syncthreads();
    }

    // pack A fragments (f16) into LDS
    {
        _Float16* aH = (_Float16*)aLds;
        for (int i = tid; i < AP_TILE; i += 256) {
            int e = i & 15, lane = (i >> 4) & 31, kt = i >> 9;
            int K = kt * 32 + a_kk(e, lane);
            int M = lane & 15;
            aH[i] = (_Float16)xLds[M * DDIM + K];
        }
    }
    __syncthreads();

    // main WMMA loop: each wave owns 64 output columns (4 tiles), K = 512
    const int wv = tid >> 5, lane = tid & 31;
    const int ct0 = wv * 4;
    v8f acc[4] = {};
    const v16h* B = (const v16h*)wt;
    for (int kt = 0; kt < KT_N; ++kt) {
        v16h a  = aLds[kt * 32 + lane];
        v16h b0 = B[(kt * 32 + ct0 + 0) * 32 + lane];
        v16h b1 = B[(kt * 32 + ct0 + 1) * 32 + lane];
        v16h b2 = B[(kt * 32 + ct0 + 2) * 32 + lane];
        v16h b3 = B[(kt * 32 + ct0 + 3) * 32 + lane];
        acc[0] = __builtin_amdgcn_wmma_f32_16x16x32_f16(false, a, false, b0, (short)0, acc[0], false, false);
        acc[1] = __builtin_amdgcn_wmma_f32_16x16x32_f16(false, a, false, b1, (short)0, acc[1], false, false);
        acc[2] = __builtin_amdgcn_wmma_f32_16x16x32_f16(false, a, false, b2, (short)0, acc[2], false, false);
        acc[3] = __builtin_amdgcn_wmma_f32_16x16x32_f16(false, a, false, b3, (short)0, acc[3], false, false);
    }

    // epilogue: scale by inner inv-norm, add bias, stage rows (reuse xLds)
    {
        int mb = (lane >= 16) ? 8 : 0;
        int nl = lane & 15;
        for (int t = 0; t < 4; ++t) {
            int N = wv * 64 + t * 16 + nl;
            float bb = bLds[N];
            for (int r = 0; r < 8; ++r) {
                int M = mb + r;
                xLds[M * DDIM + N] = acc[t][r] * invn[M] + bb;
            }
        }
    }
    __syncthreads();

    // outer l2 norm per row
    {
        int row = tid >> 4, lp = tid & 15;
        float s = 0.f;
        for (int c = lp; c < DDIM; c += 16) { float f = xLds[row * DDIM + c]; s += f * f; }
        red[tid] = s;
        __syncthreads();
        if (lp == 0) {
            float t = 0.f;
            for (int q = 0; q < 16; ++q) t += red[(row << 4) + q];
            oinv[row] = 1.0f / fmaxf(sqrtf(t), 1e-12f);
        }
        __syncthreads();
    }

    // store normalized projection in packed WMMA-A layout (f16)
    {
        _Float16* dst = proj + (size_t)(mag * TILES + tile) * AP_TILE;
        for (int i = tid; i < AP_TILE; i += 256) {
            int e = i & 15, ln = (i >> 4) & 31, kt = i >> 9;
            int K = kt * 32 + a_kk(e, ln);
            int M = ln & 15;
            dst[i] = (_Float16)(xLds[M * DDIM + K] * oinv[M]);
        }
    }
}

// ---------------- patch-descriptor scores -> patch max ----------------
__global__ __launch_bounds__(256) void score_kernel(
    const _Float16* __restrict__ proj, const _Float16* __restrict__ dp,
    float* __restrict__ pmax)
{
    __shared__ float sc[2][ROWS][48];
    const int tile = blockIdx.x, tid = threadIdx.x;
    const int wv = tid >> 5, lane = tid & 31;
    const int mag = wv >> 2, ct = wv & 3;
    if (ct < 3) {
        const v16h* A  = (const v16h*)(proj + (size_t)(mag * TILES + tile) * AP_TILE);
        const v16h* Bv = (const v16h*)dp;
        v8f acc = {};
        for (int kt = 0; kt < KT_N; ++kt) {
            v16h a = A[kt * 32 + lane];
            v16h b = Bv[(kt * 3 + ct) * 32 + lane];
            acc = __builtin_amdgcn_wmma_f32_16x16x32_f16(false, a, false, b, (short)0, acc, false, false);
        }
        int N = ct * 16 + (lane & 15), mb = (lane >= 16) ? 8 : 0;
        for (int r = 0; r < 8; ++r) sc[mag][mb + r][N] = acc[r];
    }
    __syncthreads();
    if (tid < 16) {
        // classes partition the 40 descriptors: max over class-maxes == global max
        float mx = -3.4e38f;
        for (int t = 0; t < NDESC; ++t) {
            float v = 0.5f * (sc[0][tid][t] + sc[1][tid][t]);
            mx = fmaxf(mx, v);
        }
        pmax[tile * ROWS + tid] = mx;
    }
}

// ---------------- top-100 selection per batch ----------------
__global__ __launch_bounds__(1024) void topk_kernel(
    const float* __restrict__ pmax, int* __restrict__ topi)
{
    __shared__ float sv[1024];
    __shared__ int   si[1024];
    const int b = blockIdx.x, tid = threadIdx.x;
    const float* pm = pmax + b * NPB;

    float lv[49];                               // private strided chunk
    for (int q = 0; q < 49; ++q) {
        int i = q * 1024 + tid;
        lv[q] = (i < NPB) ? pm[i] : -3.4e38f;
    }
    for (int it = 0; it < TOPJ; ++it) {
        float best = -3.4e38f; int bq = -1;
        for (int q = 0; q < 49; ++q)
            if (lv[q] > best) { best = lv[q]; bq = q; }   // ascending idx -> first tie wins
        sv[tid] = best;
        si[tid] = (bq >= 0) ? (bq * 1024 + tid) : 0x7fffffff;
        __syncthreads();
        for (int s = 512; s > 0; s >>= 1) {
            if (tid < s) {
                float v2 = sv[tid + s]; int i2 = si[tid + s];
                if (v2 > sv[tid] || (v2 == sv[tid] && i2 < si[tid])) { sv[tid] = v2; si[tid] = i2; }
            }
            __syncthreads();
        }
        int win = si[0];
        if (tid == 0) topi[b * 128 + it] = b * NPB + win;
        if ((win & 1023) == tid) lv[win >> 10] = -3.4e38f;   // owner clears its slot
        __syncthreads();
    }
}

// ---------------- gather, mean, norm, classify, loss ----------------
__global__ __launch_bounds__(512) void finalize_kernel(
    const _Float16* __restrict__ proj, const int* __restrict__ topi,
    const float* __restrict__ cf, const int* __restrict__ label,
    float* __restrict__ out)
{
    __shared__ float red[DDIM];
    __shared__ float lg[NCLS];
    __shared__ float sInv;
    const int tid = threadIdx.x;

    // inverse of the packed A layout for column d
    int d = tid;
    int kt = d >> 5, kk = d & 31;
    int hi = (kk >> 3) & 1, base = kk - (hi << 3);
    int e = (base < 8) ? base : (base - 8);

    float lossSum = 0.f;
    for (int b = 0; b < 2; ++b) {
        float acc = 0.f;
        for (int j = 0; j < TOPJ; ++j) {
            int m = topi[b * 128 + j];                 // row in x_s_proj (mag 0)
            size_t idx = ((size_t)((m >> 4) * KT_N + kt) * 32 + (m & 15) + (hi << 4)) * 16 + e;
            acc += (float)proj[idx];
        }
        float mean = acc * (1.0f / TOPJ);
        red[tid] = mean * mean;
        __syncthreads();
        for (int s = 256; s > 0; s >>= 1) { if (tid < s) red[tid] += red[tid + s]; __syncthreads(); }
        if (tid == 0) sInv = 1.0f / fmaxf(sqrtf(red[0]), 1e-12f);
        __syncthreads();
        float sf = mean * sInv;                        // slide feature component d
        for (int c = 0; c < NCLS; ++c) {
            red[tid] = sf * cf[c * DDIM + tid];
            __syncthreads();
            for (int s = 256; s > 0; s >>= 1) { if (tid < s) red[tid] += red[tid + s]; __syncthreads(); }
            if (tid == 0) lg[c] = red[0];
            __syncthreads();
        }
        if (tid == 0) {
            float mx = lg[0]; int am = 0;
            for (int c = 1; c < NCLS; ++c) if (lg[c] > mx) { mx = lg[c]; am = c; }
            float se = 0.f;
            for (int c = 0; c < NCLS; ++c) se += expf(lg[c] - mx);
            float lse = logf(se);
            for (int c = 0; c < NCLS; ++c) out[b * NCLS + c] = expf(lg[c] - mx) / se;
            out[2 * NCLS + b] = (float)am;             // Y_hat
            int lb = label[b];
            lossSum += -(lg[lb] - mx - lse);
        }
        __syncthreads();
    }
    if (tid == 0) out[2 * NCLS + 2] = lossSum * 0.5f;  // loss
}

extern "C" void kernel_launch(void* const* d_in, const int* in_sizes, int n_in,
                              void* d_out, int out_size, void* d_ws, size_t ws_size,
                              hipStream_t stream)
{
    const float* xs    = (const float*)d_in[0];
    const float* xl    = (const float*)d_in[2];
    const int*   label = (const int*)d_in[4];
    const float* W     = (const float*)d_in[5];
    const float* bias  = (const float*)d_in[6];
    const float* desc  = (const float*)d_in[7];
    const float* cf    = (const float*)d_in[8];
    float* out = (float*)d_out;

    char* ws = (char*)d_ws;
    _Float16* wt   = (_Float16*)(ws + WS_WT);
    _Float16* dp   = (_Float16*)(ws + WS_DESC);
    _Float16* proj = (_Float16*)(ws + WS_PROJ);
    float*    pmax = (float*)(ws + WS_PMAX);
    int*      topi = (int*)(ws + WS_TOPI);

    pack_w_kernel   <<<1024, 256, 0, stream>>>(W, wt);
    pack_desc_kernel<<<96,   256, 0, stream>>>(desc, dp);
    proj_gemm_kernel<<<dim3(TILES, 2), 256, 0, stream>>>(xs, xl, bias, wt, proj);
    score_kernel    <<<TILES, 256, 0, stream>>>(proj, dp, pmax);
    topk_kernel     <<<2, 1024, 0, stream>>>(pmax, topi);
    finalize_kernel <<<1, 512, 0, stream>>>(proj, topi, cf, label, out);
}
